// NTXent_50379966382615
// MI455X (gfx1250) — matrix operand order repaired and checked by reading, not compiled, not run on hardware
//
#include <hip/hip_runtime.h>
#include <hip/hip_bf16.h>

// ---------------------------------------------------------------------------
// NT-Xent loss, fused for MI455X (gfx1250, wave32, WMMA bf16 16x16x32).
//   pass A: neg_i  = sum_j exp(10*s_ij) * nm_ij          (s = bf16 WMMA GEMM)
//   pass B: S_i    = sum_j pm_ij * -log(exp(10*s_ij)/neg_i + 1e-6)
//           PS_i   = sum_j pm_ij
//   out    = mean_i S_i / max(PS_i, 1)
// ---------------------------------------------------------------------------

typedef __bf16 bf16;
typedef __attribute__((ext_vector_type(16))) __bf16        v16bf;
typedef __attribute__((ext_vector_type(8)))  float         v8f;
typedef __attribute__((ext_vector_type(4)))  unsigned int  v4u;

#define NROWS 8192
#define DDIM  768
#define BM    128          // rows per workgroup tile
#define BN    128          // cols per inner column step
#define LDA   40           // padded LDS row stride (bf16 elems): 80B -> conflict-free b128
#define CSTEPS 4           // column steps per block (chunk = 512 cols)
#define CCHUNK (BN*CSTEPS)
#define ROWBLKS (NROWS/BM)     // 64
#define COLBLKS (NROWS/CCHUNK) // 16

union BFrag { v16bf v; v4u q[2]; };

// ---------------------------------------------------------------------------
// 128x128 bf16 WMMA tile: 8 waves, each wave computes 64x32 (4x2 subtiles).
// Features staged via LDS; A/B fragments use contiguous k = 16h + s in both operands.
// ---------------------------------------------------------------------------
__device__ __forceinline__ void gemm_tile(const bf16* __restrict__ fnb, int i0, int j0,
                                          bf16* Abuf, bf16* Bbuf, v8f acc[4][2])
{
    const int tid  = threadIdx.x;
    const int lane = tid & 31;
    const int wv   = tid >> 5;
    const int wm   = (wv & 1) * 64;   // wave M offset in block tile
    const int wn   = (wv >> 1) * 32;  // wave N offset in block tile
    const int ln   = lane & 15;
    const int h    = lane >> 4;
    const int lrow = tid >> 1;        // cooperative-fill row (0..127)
    const int half = tid & 1;         // which 16-elem half of the 32-wide K slab

    for (int k0 = 0; k0 < DDIM; k0 += 32) {
        // global -> regs (16B x2 per buffer per thread)
        const v4u* ga = (const v4u*)(fnb + (size_t)(i0 + lrow) * DDIM + k0 + half * 16);
        const v4u* gb = (const v4u*)(fnb + (size_t)(j0 + lrow) * DDIM + k0 + half * 16);
        v4u va0 = ga[0], va1 = ga[1];
        v4u vb0 = gb[0], vb1 = gb[1];

        __syncthreads();              // previous slab fully consumed
        v4u* da = (v4u*)(Abuf + lrow * LDA + half * 16);
        v4u* db = (v4u*)(Bbuf + lrow * LDA + half * 16);
        da[0] = va0; da[1] = va1;
        db[0] = vb0; db[1] = vb1;
        __syncthreads();

        // B fragments: 16 contiguous bf16 of row (wn+ni*16+ln) at k-offset h*16
        BFrag b0, b1;
        {
            const bf16* bp0 = Bbuf + (wn + 0 * 16 + ln) * LDA + h * 16;
            const bf16* bp1 = Bbuf + (wn + 1 * 16 + ln) * LDA + h * 16;
            b0.q[0] = *(const v4u*)bp0;  b0.q[1] = *(const v4u*)(bp0 + 8);
            b1.q[0] = *(const v4u*)bp1;  b1.q[1] = *(const v4u*)(bp1 + 8);
        }
        #pragma unroll
        for (int mi = 0; mi < 4; ++mi) {
            // A fragment: 16 contiguous bf16 [kbase..kbase+15], kbase=h*16
            const bf16* ap = Abuf + (wm + mi * 16 + ln) * LDA + h * 16;
            BFrag a;
            a.q[0] = *(const v4u*)ap;
            a.q[1] = *(const v4u*)(ap + 8);
            acc[mi][0] = __builtin_amdgcn_wmma_f32_16x16x32_bf16(
                false, a.v, false, b0.v, (short)0, acc[mi][0], false, false);
            acc[mi][1] = __builtin_amdgcn_wmma_f32_16x16x32_bf16(
                false, a.v, false, b1.v, (short)0, acc[mi][1], false, false);
        }
    }
}

// ---------------------------------------------------------------------------
// Kernel 1: row-normalize features -> bf16
// ---------------------------------------------------------------------------
__global__ void __launch_bounds__(256) k_norm(const float* __restrict__ f,
                                              bf16* __restrict__ fnb)
{
    const int row = blockIdx.x;
    const int t   = threadIdx.x;
    const float* fr = f + (size_t)row * DDIM;
    float v0 = fr[t], v1 = fr[t + 256], v2 = fr[t + 512];
    float ss = v0 * v0 + v1 * v1 + v2 * v2;
    ss += __shfl_xor(ss, 1);  ss += __shfl_xor(ss, 2);
    ss += __shfl_xor(ss, 4);  ss += __shfl_xor(ss, 8);
    ss += __shfl_xor(ss, 16);

    __shared__ float wred[8];
    __shared__ float rn_s;
    if ((t & 31) == 0) wred[t >> 5] = ss;
    __syncthreads();
    if (t == 0) {
        float tot = 0.f;
        #pragma unroll
        for (int w = 0; w < 8; ++w) tot += wred[w];
        rn_s = 1.f / fmaxf(sqrtf(tot), 1e-8f);
    }
    __syncthreads();
    const float rn = rn_s;
    bf16* o = fnb + (size_t)row * DDIM;
    o[t]       = (bf16)(v0 * rn);
    o[t + 256] = (bf16)(v1 * rn);
    o[t + 512] = (bf16)(v2 * rn);
}

// ---------------------------------------------------------------------------
// Kernel 2: neg_i = sum_j exp(10*s_ij) * nm_ij   (diag excluded)
// ---------------------------------------------------------------------------
__global__ void __launch_bounds__(256) k_neg(const bf16* __restrict__ fnb,
                                             const float* __restrict__ nm,
                                             float* __restrict__ neg)
{
    __shared__ __align__(16) bf16 Abuf[BM * LDA];
    __shared__ __align__(16) bf16 Bbuf[BN * LDA];

    const int i0  = blockIdx.x * BM;
    const int jc  = blockIdx.y * CCHUNK;
    const int tid = threadIdx.x, lane = tid & 31, wv = tid >> 5;
    const int wm  = (wv & 1) * 64, wn = (wv >> 1) * 32;
    const int ln  = lane & 15, h = lane >> 4;

    float rowsum[4][8];
    #pragma unroll
    for (int mi = 0; mi < 4; ++mi)
        #pragma unroll
        for (int r = 0; r < 8; ++r) rowsum[mi][r] = 0.f;

    for (int cs = 0; cs < CSTEPS; ++cs) {
        const int j0 = jc + cs * BN;
        v8f acc[4][2];
        #pragma unroll
        for (int mi = 0; mi < 4; ++mi) { acc[mi][0] = (v8f)0.f; acc[mi][1] = (v8f)0.f; }

        gemm_tile(fnb, i0, j0, Abuf, Bbuf, acc);

        #pragma unroll
        for (int mi = 0; mi < 4; ++mi) {
            #pragma unroll
            for (int r = 0; r < 8; ++r) {
                const int i = i0 + wm + mi * 16 + r + 8 * h;
                float s = 0.f;
                #pragma unroll
                for (int ni = 0; ni < 2; ++ni) {
                    const int j = j0 + wn + ni * 16 + ln;
                    float m = nm[(size_t)i * NROWS + j];
                    if (i == j) m = 0.f;
                    s += __expf(10.f * acc[mi][ni][r]) * m;
                }
                rowsum[mi][r] += s;
            }
        }
    }
    #pragma unroll
    for (int mi = 0; mi < 4; ++mi) {
        #pragma unroll
        for (int r = 0; r < 8; ++r) {
            float v = rowsum[mi][r];
            v += __shfl_xor(v, 1); v += __shfl_xor(v, 2);
            v += __shfl_xor(v, 4); v += __shfl_xor(v, 8);
            if (ln == 0) atomicAdd(&neg[i0 + wm + mi * 16 + r + 8 * h], v);
        }
    }
}

// ---------------------------------------------------------------------------
// Kernel 3: S_i = sum_j pm_ij * -log(exp(10*s_ij)/neg_i + 1e-6); PS_i = sum pm
// ---------------------------------------------------------------------------
__global__ void __launch_bounds__(256) k_loss(const bf16* __restrict__ fnb,
                                              const float* __restrict__ pm,
                                              const float* __restrict__ neg,
                                              float* __restrict__ S,
                                              float* __restrict__ PS)
{
    __shared__ __align__(16) bf16 Abuf[BM * LDA];
    __shared__ __align__(16) bf16 Bbuf[BN * LDA];

    const int i0  = blockIdx.x * BM;
    const int jc  = blockIdx.y * CCHUNK;
    const int tid = threadIdx.x, lane = tid & 31, wv = tid >> 5;
    const int wm  = (wv & 1) * 64, wn = (wv >> 1) * 32;
    const int ln  = lane & 15, h = lane >> 4;

    float rneg[4][8], Ssum[4][8], Pc[4][8];
    #pragma unroll
    for (int mi = 0; mi < 4; ++mi)
        #pragma unroll
        for (int r = 0; r < 8; ++r) {
            rneg[mi][r] = 1.0f / neg[i0 + wm + mi * 16 + r + 8 * h];
            Ssum[mi][r] = 0.f;
            Pc[mi][r]   = 0.f;
        }

    for (int cs = 0; cs < CSTEPS; ++cs) {
        const int j0 = jc + cs * BN;
        v8f acc[4][2];
        #pragma unroll
        for (int mi = 0; mi < 4; ++mi) { acc[mi][0] = (v8f)0.f; acc[mi][1] = (v8f)0.f; }

        gemm_tile(fnb, i0, j0, Abuf, Bbuf, acc);

        #pragma unroll
        for (int mi = 0; mi < 4; ++mi) {
            #pragma unroll
            for (int r = 0; r < 8; ++r) {
                const int i = i0 + wm + mi * 16 + r + 8 * h;
                float sacc = 0.f, pacc = 0.f;
                #pragma unroll
                for (int ni = 0; ni < 2; ++ni) {
                    const int j = j0 + wn + ni * 16 + ln;
                    float p = pm[(size_t)i * NROWS + j];
                    if (i == j) p = 0.f;
                    float sim = __expf(10.f * acc[mi][ni][r]);
                    float lp  = -__logf(sim * rneg[mi][r] + 1e-6f);
                    sacc += p * lp;
                    pacc += p;
                }
                Ssum[mi][r] += sacc;
                Pc[mi][r]   += pacc;
            }
        }
    }
    #pragma unroll
    for (int mi = 0; mi < 4; ++mi) {
        #pragma unroll
        for (int r = 0; r < 8; ++r) {
            float v = Ssum[mi][r];
            v += __shfl_xor(v, 1); v += __shfl_xor(v, 2);
            v += __shfl_xor(v, 4); v += __shfl_xor(v, 8);
            float q = Pc[mi][r];
            q += __shfl_xor(q, 1); q += __shfl_xor(q, 2);
            q += __shfl_xor(q, 4); q += __shfl_xor(q, 8);
            if (ln == 0) {
                const int i = i0 + wm + mi * 16 + r + 8 * h;
                atomicAdd(&S[i], v);
                atomicAdd(&PS[i], q);
            }
        }
    }
}

// ---------------------------------------------------------------------------
// Kernel 4: out = mean_i S_i / max(PS_i, 1)
// ---------------------------------------------------------------------------
__global__ void __launch_bounds__(256) k_final(const float* __restrict__ S,
                                               const float* __restrict__ PS,
                                               float* __restrict__ out)
{
    __shared__ float red[256];
    float acc = 0.f;
    for (int i = threadIdx.x; i < NROWS; i += 256) {
        float ps = PS[i];
        ps = (ps < 0.5f) ? 1.f : ps;
        acc += S[i] / ps;
    }
    red[threadIdx.x] = acc;
    __syncthreads();
    for (int s = 128; s > 0; s >>= 1) {
        if (threadIdx.x < s) red[threadIdx.x] += red[threadIdx.x + s];
        __syncthreads();
    }
    if (threadIdx.x == 0) out[0] = red[0] / (float)NROWS;
}

// ---------------------------------------------------------------------------
extern "C" void kernel_launch(void* const* d_in, const int* in_sizes, int n_in,
                              void* d_out, int out_size, void* d_ws, size_t ws_size,
                              hipStream_t stream)
{
    (void)in_sizes; (void)n_in; (void)out_size; (void)ws_size;
    const float* feat = (const float*)d_in[0];
    const float* pm   = (const float*)d_in[1];
    const float* nm   = (const float*)d_in[2];

    char* ws  = (char*)d_ws;
    bf16* fnb = (bf16*)ws;
    size_t off = (size_t)NROWS * DDIM * sizeof(bf16);   // 12.6 MB
    float* neg = (float*)(ws + off);
    float* S   = neg + NROWS;
    float* PS  = S + NROWS;

    hipMemsetAsync(neg, 0, 3 * NROWS * sizeof(float), stream);
    k_norm<<<NROWS, 256, 0, stream>>>(feat, fnb);
    k_neg<<<dim3(ROWBLKS, COLBLKS), 256, 0, stream>>>(fnb, nm, neg);
    k_loss<<<dim3(ROWBLKS, COLBLKS), 256, 0, stream>>>(fnb, pm, neg, S, PS);
    k_final<<<1, 256, 0, stream>>>(S, PS, (float*)d_out);
}
